// Net_44083544326216
// MI455X (gfx1250) — matrix operand order repaired and checked by reading, not compiled, not run on hardware
//
#include <hip/hip_runtime.h>
#include <hip/hip_bf16.h>

// ---------------------------------------------------------------------------
// MI455X (gfx1250) implementation of the SplineConv GNN reference.
// Dense math (kernel transforms, root weights, 1x1 shortcut, fc1) runs through
// a wave32 WMMA f16 GEMM (v_wmma_f32_16x16x32_f16) fed by async global->LDS
// copies (ASYNCcnt path). All GEMM operands are padded to tile-exact shapes
// (K->mult of 32 with zeros, N->mult of 128 with zero weight rows, M via
// buffer capacity), so the GEMM has no guards and no slow path.
// ---------------------------------------------------------------------------

typedef _Float16 half_t;
typedef __attribute__((ext_vector_type(16))) _Float16 v16h;
typedef __attribute__((ext_vector_type(8)))  float    v8f;
typedef __attribute__((ext_vector_type(4)))  int      int4v;

#define NN      2048      // nodes
#define EE      32768     // edges
#define KD      125       // 5^3 spline kernels
#define KDP     128       // padded kernel count (zero rows 125..127)
#define KC      8         // kernel chunk size (16 uniform chunks of 8)
#define NGRAPH  8
#define POOLC   64
#define NSEG    (NGRAPH * POOLC)
#define NCLS    10
#define FC1IN   (POOLC * 512)   // 32768
#define FC1OUT  1024

// ---- CDNA5 async global->LDS copy (ASYNCcnt path), with safe fallback ------
#if defined(__has_builtin)
# if __has_builtin(__builtin_amdgcn_global_load_async_to_lds_b128) && \
     __has_builtin(__builtin_amdgcn_s_wait_asynccnt)
#  define USE_ASYNC_LDS 1
# endif
#endif
#ifndef USE_ASYNC_LDS
# define USE_ASYNC_LDS 0
#endif

__device__ __forceinline__ void copy16_to_lds(const half_t* g, half_t* l) {
#if USE_ASYNC_LDS
  __builtin_amdgcn_global_load_async_to_lds_b128(
      (__attribute__((address_space(1))) int4v*)(g),
      (__attribute__((address_space(3))) int4v*)(l), 0, 0);
#else
  *(uint4*)l = *(const uint4*)g;
#endif
}
__device__ __forceinline__ void async_wait() {
#if USE_ASYNC_LDS
  __builtin_amdgcn_s_wait_asynccnt(0);
#endif
}

// ---------------------------------------------------------------------------
// WMMA GEMM: C[M,N] = A[M,K] * B[K,N] with B supplied TRANSPOSED:
//   BT is [N][K] row-major (leading dim ldbt). A is [M][K] f16 row-major.
// REQUIREMENTS (guaranteed by callers via padding):
//   K % 32 == 0, N % 128 == 0, and A/C have >= ceil(M/64)*64 rows of capacity.
// Block: 256 threads = 8 wave32 waves; tile 64(M) x 128(N), K-step 32.
// Wave (wm 0..1, wn 0..3) owns a 32x32 patch -> 4 WMMAs/K-step with reuse.
// ---------------------------------------------------------------------------
#define BM 64
#define BN 128
#define BK 32
#define APAD 8

__global__ __launch_bounds__(256)
void wmma_gemm(const half_t* __restrict__ A, const half_t* __restrict__ BT,
               float* __restrict__ C, int K, int lda, int ldbt, int ldc)
{
  __shared__ half_t As[BM][BK + APAD];          // [m][k]
  __shared__ half_t Bs[BN][BK + APAD];          // [n][k]  (B^T tile)

  const int tid  = threadIdx.x;
  const int wave = tid >> 5;
  const int lane = tid & 31;
  const int hlf  = lane >> 4;                   // 0 or 1
  const int l16  = lane & 15;

  const int bm = blockIdx.y * BM;
  const int bn = blockIdx.x * BN;
  const int wm = wave >> 2;                     // 0..1 -> 32-row strip
  const int wn = wave & 3;                      // 0..3 -> 32-col strip

  const int ar = tid >> 2;                      // 0..63  staging row
  const int ac = (tid & 3) * 8;                 // 0/8/16/24 staging col (halfs)

  const half_t* aG  = &A [(long)(bm + ar)      * lda  + ac];
  const half_t* bG0 = &BT[(long)(bn + ar)      * ldbt + ac];
  const half_t* bG1 = &BT[(long)(bn + 64 + ar) * ldbt + ac];

  v8f acc00 = {}, acc01 = {}, acc10 = {}, acc11 = {};

  for (int k0 = 0; k0 < K; k0 += BK) {
    copy16_to_lds(aG  + k0, &As[ar][ac]);
    copy16_to_lds(bG0 + k0, &Bs[ar][ac]);
    copy16_to_lds(bG1 + k0, &Bs[64 + ar][ac]);
    if (k0 + BK < K) {
      __builtin_prefetch(aG  + k0 + BK, 0, 1);
      __builtin_prefetch(bG0 + k0 + BK, 0, 1);
      __builtin_prefetch(bG1 + k0 + BK, 0, 1);
    }
    async_wait();
    __syncthreads();

    // ---- fragments (16-bit A 16x32 layout; B^T rows give B columns)
    v16h a0, a1, b0, b1;
    {
      const half_t* p = &As[wm * 32 + l16][0];
      #pragma unroll
      for (int j = 0; j < 8; ++j) { a0[j] = p[8 * hlf + j]; a0[8 + j] = p[16 + 8 * hlf + j]; }
      p = &As[wm * 32 + 16 + l16][0];
      #pragma unroll
      for (int j = 0; j < 8; ++j) { a1[j] = p[8 * hlf + j]; a1[8 + j] = p[16 + 8 * hlf + j]; }
    }
    {
      const half_t* p = &Bs[wn * 32 + l16][0];
      #pragma unroll
      for (int j = 0; j < 8; ++j) { b0[j] = p[8 * hlf + j]; b0[8 + j] = p[16 + 8 * hlf + j]; }
      p = &Bs[wn * 32 + 16 + l16][0];
      #pragma unroll
      for (int j = 0; j < 8; ++j) { b1[j] = p[8 * hlf + j]; b1[8 + j] = p[16 + 8 * hlf + j]; }
    }
    acc00 = __builtin_amdgcn_wmma_f32_16x16x32_f16(false, a0, false, b0, (short)0, acc00, false, false);
    acc01 = __builtin_amdgcn_wmma_f32_16x16x32_f16(false, a0, false, b1, (short)0, acc01, false, false);
    acc10 = __builtin_amdgcn_wmma_f32_16x16x32_f16(false, a1, false, b0, (short)0, acc10, false, false);
    acc11 = __builtin_amdgcn_wmma_f32_16x16x32_f16(false, a1, false, b1, (short)0, acc11, false, false);
    __syncthreads();
  }

  // ---- store C per the 16x16 f32 layout: VGPR j -> row j + 8*hlf
  int n0 = bn + wn * 32 + l16;
  int n1 = n0 + 16;
  #pragma unroll
  for (int j = 0; j < 8; ++j) {
    int r0 = bm + wm * 32 + j + hlf * 8;
    int r1 = r0 + 16;
    C[(long)r0 * ldc + n0] = acc00[j];
    C[(long)r0 * ldc + n1] = acc01[j];
    C[(long)r1 * ldc + n0] = acc10[j];
    C[(long)r1 * ldc + n1] = acc11[j];
  }
}

// ---------------------------------------------------------------------------
// Cast / pack kernels (all produce tile-exact padded operands)
// ---------------------------------------------------------------------------
__global__ void k_zero_f32(float* p, long n) {
  long i = (long)blockIdx.x * blockDim.x + threadIdx.x;
  if (i < n) p[i] = 0.f;
}

__global__ void k_cast_f32_f16(const float* __restrict__ in, half_t* __restrict__ out, long n) {
  long i = (long)blockIdx.x * blockDim.x + threadIdx.x;
  if (i < n) out[i] = (half_t)in[i];
}

// in [n][cin] f32 -> out [n][cinp] f16 with zero columns cin..cinp
__global__ void k_cast_pad(const float* __restrict__ in, half_t* __restrict__ out,
                           int n, int cin, int cinp) {
  long total = (long)n * cinp;
  long i = (long)blockIdx.x * blockDim.x + threadIdx.x;
  if (i >= total) return;
  int c = (int)(i % cinp);
  int r = (int)(i / cinp);
  out[i] = (c < cin) ? (half_t)in[(long)r * cin + c] : (half_t)0.f;
}

// in [rows][cols] f32 -> out [colsp][rowsp] f16 (transposed, zero padded)
__global__ void k_castT_pad(const float* __restrict__ in, half_t* __restrict__ out,
                            int rows, int cols, int rowsp, int colsp) {
  long total = (long)rowsp * colsp;
  long i = (long)blockIdx.x * blockDim.x + threadIdx.x;
  if (i >= total) return;
  int r = (int)(i % rowsp);   // K index
  int c = (int)(i / rowsp);   // N index
  out[i] = (r < rows && c < cols) ? (half_t)in[(long)r * cols + c] : (half_t)0.f;
}

// W[kd][cin][cout] f32 -> Wt[kdp*cout][cinp] f16 (B^T rows (k,co), zero padded)
__global__ void k_pack_wT(const float* __restrict__ w, half_t* __restrict__ out,
                          int kd, int kdp, int cin, int cinp, int cout) {
  long total = (long)kdp * cout * cinp;
  long i = (long)blockIdx.x * blockDim.x + threadIdx.x;
  if (i >= total) return;
  int  ci  = (int)(i % cinp);
  long row = i / cinp;                 // k*cout + co
  int  co  = (int)(row % cout);
  int  k   = (int)(row / cout);
  out[i] = (k < kd && ci < cin) ? (half_t)w[((long)k * cin + ci) * cout + co] : (half_t)0.f;
}

__global__ void k_deg(const int* __restrict__ dst, float* __restrict__ deg, int e) {
  int i = blockIdx.x * blockDim.x + threadIdx.x;
  if (i < e) atomicAdd(&deg[dst[i]], 1.0f);
}

// ---------------------------------------------------------------------------
// Edge pass for spline conv: gather Yc rows with B-spline basis weights.
// Yc = [NN, KC*Cout] for kernels [kbase, kbase+KC). msg accumulates per edge.
// ---------------------------------------------------------------------------
__global__ __launch_bounds__(128)
void k_edge_spline(const float* __restrict__ eattr, const int* __restrict__ src,
                   const float* __restrict__ Yc, float* __restrict__ msg,
                   int cout, int kbase)
{
  int e = blockIdx.x;
  float v0 = eattr[3 * e + 0] * 4.0f;
  float v1 = eattr[3 * e + 1] * 4.0f;
  float v2 = eattr[3 * e + 2] * 4.0f;
  float b0 = fmaxf(fminf(floorf(v0), 3.f), 0.f);
  float b1 = fmaxf(fminf(floorf(v1), 3.f), 0.f);
  float b2 = fmaxf(fminf(floorf(v2), 3.f), 0.f);
  float f0 = v0 - b0, f1 = v1 - b1, f2 = v2 - b2;
  float w0[2] = {1.f - f0, f0};
  float w1[2] = {1.f - f1, f1};
  float w2[2] = {1.f - f2, f2};
  int   i0[2] = {(int)b0, (int)b0 + 1};
  int   i1[2] = {(int)b1, (int)b1 + 1};
  int   i2[2] = {(int)b2, (int)b2 + 1};

  int kid[8]; float kw[8]; int nact = 0;
  #pragma unroll
  for (int a = 0; a < 2; ++a)
    #pragma unroll
    for (int b = 0; b < 2; ++b)
      #pragma unroll
      for (int c = 0; c < 2; ++c) {
        int k = (i0[a] * 5 + i1[b]) * 5 + i2[c] - kbase;
        if (k >= 0 && k < KC) { kid[nact] = k; kw[nact] = w0[a] * w1[b] * w2[c]; ++nact; }
      }
  if (nact == 0) return;

  long yb = (long)src[e] * ((long)KC * cout);
  long mb = (long)e * cout;
  for (int ch = threadIdx.x; ch < cout; ch += blockDim.x) {
    float acc = 0.f;
    for (int s = 0; s < nact; ++s)
      acc += kw[s] * Yc[yb + (long)kid[s] * cout + ch];
    msg[mb + ch] += acc;
  }
}

// agg[dst[e]] += msg[e]  (segment sum over edges)
__global__ __launch_bounds__(128)
void k_scatter_msg(const float* __restrict__ msg, const int* __restrict__ dst,
                   float* __restrict__ agg, int cout) {
  int e = blockIdx.x;
  long mb = (long)e * cout;
  long ab = (long)dst[e] * cout;
  for (int ch = threadIdx.x; ch < cout; ch += blockDim.x)
    atomicAdd(&agg[ab + ch], msg[mb + ch]);
}

// 1x1 conv edge pass: agg[dst[e]] += Y1[src[e]]   (Y1 has row stride ldy)
__global__ __launch_bounds__(128)
void k_scatter_rows(const float* __restrict__ Y1, const int* __restrict__ src,
                    const int* __restrict__ dst, float* __restrict__ agg,
                    int cout, int ldy) {
  int e = blockIdx.x;
  long yb = (long)src[e] * ldy;
  long ab = (long)dst[e] * cout;
  for (int ch = threadIdx.x; ch < cout; ch += blockDim.x)
    atomicAdd(&agg[ab + ch], Y1[yb + ch]);
}

// out = agg/max(deg,1) + rootout + bias   (rootout has row stride ldr)
__global__ void k_finish_conv(const float* __restrict__ agg, const float* __restrict__ deg,
                              const float* __restrict__ rootout, const float* __restrict__ bias,
                              float* __restrict__ out, int cout, int ldr) {
  long i = (long)blockIdx.x * blockDim.x + threadIdx.x;
  long tot = (long)NN * cout;
  if (i >= tot) return;
  int n = (int)(i / cout), c = (int)(i % cout);
  out[i] = agg[i] / fmaxf(deg[n], 1.f) + rootout[(long)n * ldr + c] + bias[c];
}

__global__ void k_elu_inplace(float* x, long n) {
  long i = (long)blockIdx.x * blockDim.x + threadIdx.x;
  if (i < n) { float v = x[i]; x[i] = v > 0.f ? v : expm1f(v); }
}

__global__ __launch_bounds__(256)
void k_bn_stats(const float* __restrict__ x, int rows, int cout,
                float* __restrict__ mu, float* __restrict__ rsig) {
  int c = blockIdx.x;
  float s = 0.f, s2 = 0.f;
  for (int n = threadIdx.x; n < rows; n += blockDim.x) {
    float v = x[(long)n * cout + c];
    s += v; s2 += v * v;
  }
  __shared__ float sh[256], sh2[256];
  sh[threadIdx.x] = s; sh2[threadIdx.x] = s2;
  __syncthreads();
  for (int o = 128; o > 0; o >>= 1) {
    if ((int)threadIdx.x < o) { sh[threadIdx.x] += sh[threadIdx.x + o]; sh2[threadIdx.x] += sh2[threadIdx.x + o]; }
    __syncthreads();
  }
  if (threadIdx.x == 0) {
    float m = sh[0] / rows;
    float var = sh2[0] / rows - m * m;
    mu[c] = m;
    rsig[c] = rsqrtf(var + 1e-5f);
  }
}

__global__ void k_bn_apply(const float* __restrict__ x, float* __restrict__ y,
                           const float* __restrict__ g, const float* __restrict__ b,
                           const float* __restrict__ mu, const float* __restrict__ rsig,
                           int cout, int do_elu) {
  long i = (long)blockIdx.x * blockDim.x + threadIdx.x;
  long tot = (long)NN * cout;
  if (i >= tot) return;
  int c = (int)(i % cout);
  float v = g[c] * (x[i] - mu[c]) * rsig[c] + b[c];
  y[i] = (do_elu && v <= 0.f) ? expm1f(v) : v;
}

__global__ void k_add_elu(const float* a, const float* b, float* out, long n) {
  long i = (long)blockIdx.x * blockDim.x + threadIdx.x;
  if (i < n) { float v = a[i] + b[i]; out[i] = v > 0.f ? v : expm1f(v); }
}

// ---------------------------------------------------------------------------
// Pooling + head
// ---------------------------------------------------------------------------
__global__ void k_seg(const float* __restrict__ pos, const int* __restrict__ batch,
                      int* __restrict__ seg) {
  int n = blockIdx.x * blockDim.x + threadIdx.x;
  if (n >= NN) return;
  int cx = (int)floorf(pos[3 * n + 0] / 32.0f);
  int cy = (int)floorf(pos[3 * n + 1] / 32.0f);
  seg[n] = batch[n] * POOLC + ((cx * 9 + cy) % POOLC);
}

__global__ __launch_bounds__(256)
void k_pool_max(const float* __restrict__ x, const int* __restrict__ seg,
                float* __restrict__ pooled, int cout) {
  __shared__ int list[256];
  __shared__ int cnt;
  int sid = blockIdx.x;
  if (threadIdx.x == 0) cnt = 0;
  __syncthreads();
  for (int n = threadIdx.x; n < NN; n += blockDim.x)
    if (seg[n] == sid) { int i = atomicAdd(&cnt, 1); if (i < 256) list[i] = n; }
  __syncthreads();
  int m = cnt < 256 ? cnt : 256;
  for (int c = threadIdx.x; c < cout; c += blockDim.x) {
    float mx = -INFINITY;
    for (int i = 0; i < m; ++i) mx = fmaxf(mx, x[(long)list[i] * cout + c]);
    pooled[(long)sid * cout + c] = isfinite(mx) ? mx : 0.f;
  }
}

__global__ void k_bias_elu(float* x, const float* b, int cols, long n) {
  long i = (long)blockIdx.x * blockDim.x + threadIdx.x;
  if (i < n) {
    float v = x[i] + b[i % cols];
    x[i] = v > 0.f ? v : expm1f(v);
  }
}

__global__ __launch_bounds__(32)
void k_fc2_logsoftmax(const float* __restrict__ h, const float* __restrict__ W,
                      const float* __restrict__ b, float* __restrict__ out) {
  int g = blockIdx.x;
  __shared__ float logits[NCLS];
  for (int cls = threadIdx.x; cls < NCLS; cls += blockDim.x) {
    float s = b[cls];
    for (int k = 0; k < FC1OUT; ++k) s += h[(long)g * FC1OUT + k] * W[(long)k * NCLS + cls];
    logits[cls] = s;
  }
  __syncthreads();
  if (threadIdx.x == 0) {
    float mx = -INFINITY;
    for (int j = 0; j < NCLS; ++j) mx = fmaxf(mx, logits[j]);
    float se = 0.f;
    for (int j = 0; j < NCLS; ++j) se += expf(logits[j] - mx);
    float lse = logf(se) + mx;
    for (int j = 0; j < NCLS; ++j) out[g * NCLS + j] = logits[j] - lse;
  }
}

// ---------------------------------------------------------------------------
// Host orchestration
// ---------------------------------------------------------------------------
struct Ws {
  half_t *Xh, *Wh, *Rh, *pooledh;
  float  *Yc, *msg, *agg, *rootout, *deg;
  float  *xa, *tb, *hb, *sb, *mu, *rsig, *pooled, *fc1out;
  int    *seg;
};

static inline long cdiv(long a, long b) { return (a + b - 1) / b; }

static void zero_f32(hipStream_t s, float* p, long n) {
  k_zero_f32<<<dim3((unsigned)cdiv(n, 256)), 256, 0, s>>>(p, n);
}
// N % 128 == 0, K % 32 == 0, A/C row capacity >= ceil(M/64)*64
static void gemm(hipStream_t s, const half_t* A, const half_t* BT, float* C,
                 int M, int N, int K, int lda, int ldbt, int ldc) {
  dim3 grid((unsigned)(N / BN), (unsigned)cdiv(M, BM));
  wmma_gemm<<<grid, 256, 0, s>>>(A, BT, C, K, lda, ldbt, ldc);
}

// full spline conv (kernel dim padded 125 -> 128, uniform chunks of 8)
static void spline_conv(hipStream_t s, const Ws& w,
                        const float* xin, int cin, int cout,
                        const float* Wp, const float* rootp, const float* biasp,
                        const int* src, const int* dst, const float* eattr,
                        float* out) {
  const int cinp  = (cin  < 32)  ? 32  : cin;    // K padded to >= 32
  const int coutp = (cout < 128) ? 128 : cout;   // N padded to >= 128
  k_cast_pad<<<dim3((unsigned)cdiv((long)NN * cinp, 256)), 256, 0, s>>>(xin, w.Xh, NN, cin, cinp);
  long wtot = (long)KDP * cout * cinp;
  k_pack_wT<<<dim3((unsigned)cdiv(wtot, 256)), 256, 0, s>>>(Wp, w.Wh, KD, KDP, cin, cinp, cout);
  zero_f32(s, w.msg, (long)EE * cout);
  for (int base = 0; base < KDP; base += KC) {
    gemm(s, w.Xh, w.Wh + (long)base * cout * cinp, w.Yc,
         NN, KC * cout, cinp, cinp, cinp, KC * cout);
    k_edge_spline<<<dim3(EE), 128, 0, s>>>(eattr, src, w.Yc, w.msg, cout, base);
  }
  zero_f32(s, w.agg, (long)NN * cout);
  k_scatter_msg<<<dim3(EE), 128, 0, s>>>(w.msg, dst, w.agg, cout);
  k_castT_pad<<<dim3((unsigned)cdiv((long)cinp * coutp, 256)), 256, 0, s>>>(rootp, w.Rh, cin, cout, cinp, coutp);
  gemm(s, w.Xh, w.Rh, w.rootout, NN, coutp, cinp, cinp, cinp, coutp);
  long tot = (long)NN * cout;
  k_finish_conv<<<dim3((unsigned)cdiv(tot, 256)), 256, 0, s>>>(w.agg, w.deg, w.rootout, biasp, out, cout, coutp);
}

// kernel_size=1 shortcut conv
static void conv1x1(hipStream_t s, const Ws& w,
                    const float* xin, int cin, int cout,
                    const float* Wp, const float* rootp, const float* biasp,
                    const int* src, const int* dst, float* out) {
  const int cinp  = (cin  < 32)  ? 32  : cin;
  const int coutp = (cout < 128) ? 128 : cout;
  k_cast_pad<<<dim3((unsigned)cdiv((long)NN * cinp, 256)), 256, 0, s>>>(xin, w.Xh, NN, cin, cinp);
  k_castT_pad<<<dim3((unsigned)cdiv((long)cinp * coutp, 256)), 256, 0, s>>>(Wp, w.Wh, cin, cout, cinp, coutp);
  gemm(s, w.Xh, w.Wh, w.Yc, NN, coutp, cinp, cinp, cinp, coutp);
  zero_f32(s, w.agg, (long)NN * cout);
  k_scatter_rows<<<dim3(EE), 128, 0, s>>>(w.Yc, src, dst, w.agg, cout, coutp);
  k_castT_pad<<<dim3((unsigned)cdiv((long)cinp * coutp, 256)), 256, 0, s>>>(rootp, w.Rh, cin, cout, cinp, coutp);
  gemm(s, w.Xh, w.Rh, w.rootout, NN, coutp, cinp, cinp, cinp, coutp);
  long tot = (long)NN * cout;
  k_finish_conv<<<dim3((unsigned)cdiv(tot, 256)), 256, 0, s>>>(w.agg, w.deg, w.rootout, biasp, out, cout, coutp);
}

static void bn_stats(hipStream_t s, const float* x, int cout, const Ws& w) {
  k_bn_stats<<<dim3(cout), 256, 0, s>>>(x, NN, cout, w.mu, w.rsig);
}
static void bn_apply(hipStream_t s, const float* x, float* y, const float* g,
                     const float* b, int cout, int elu, const Ws& w) {
  long tot = (long)NN * cout;
  k_bn_apply<<<dim3((unsigned)cdiv(tot, 256)), 256, 0, s>>>(x, y, g, b, w.mu, w.rsig, cout, elu);
}

extern "C" void kernel_launch(void* const* d_in, const int* in_sizes, int n_in,
                              void* d_out, int out_size, void* d_ws, size_t ws_size,
                              hipStream_t stream) {
  (void)in_sizes; (void)n_in; (void)out_size; (void)ws_size;

  // ---- inputs (setup_inputs dict order, params depth-first insertion order)
  const float* x_in  = (const float*)d_in[0];          // [N,1]
  const float* pos   = (const float*)d_in[1];          // [N,3]
  const int*   eidx  = (const int*)  d_in[2];          // [2,E]
  const float* eattr = (const float*)d_in[3];          // [E,3]
  const int*   batch = (const int*)  d_in[4];          // [N]
  const int* src = eidx;
  const int* dst = eidx + EE;

  auto P = [&](int i) { return (const float*)d_in[i]; };

  // ---- workspace layout
  char* base = (char*)d_ws;
  size_t off = 0;
  auto alloc = [&](size_t bytes) { void* p = base + off; off = (off + bytes + 255) & ~(size_t)255; return p; };
  Ws w;
  w.Xh      = (half_t*)alloc((long)NN * 512 * 2);
  w.Wh      = (half_t*)alloc((size_t)33554432 * 2);       // max(128*512*512, 32768*1024) halfs
  w.Rh      = (half_t*)alloc((long)512 * 512 * 2);
  w.pooledh = (half_t*)alloc((long)64 * FC1IN * 2);       // M padded to 64 rows
  w.Yc      = (float*)alloc((long)NN * KC * 512 * 4);
  w.msg     = (float*)alloc((long)EE * 512 * 4);
  w.agg     = (float*)alloc((long)NN * 512 * 4);
  w.rootout = (float*)alloc((long)NN * 512 * 4);
  w.deg     = (float*)alloc((long)NN * 4);
  w.xa      = (float*)alloc((long)NN * 512 * 4);
  w.tb      = (float*)alloc((long)NN * 512 * 4);
  w.hb      = (float*)alloc((long)NN * 512 * 4);
  w.sb      = (float*)alloc((long)NN * 512 * 4);
  w.mu      = (float*)alloc(512 * 4);
  w.rsig    = (float*)alloc(512 * 4);
  w.pooled  = (float*)alloc((long)NSEG * 512 * 4);
  w.fc1out  = (float*)alloc((long)64 * FC1OUT * 4);       // M padded to 64 rows
  w.seg     = (int*)alloc((long)NN * 4);

  // ---- degree (shared by all convs)
  zero_f32(stream, w.deg, NN);
  k_deg<<<dim3((unsigned)cdiv(EE, 256)), 256, 0, stream>>>(dst, w.deg, EE);

  // ---- conv1 + bn1:  x = bn(elu(spline_conv(x)))
  spline_conv(stream, w, x_in, 1, 64, P(5), P(6), P(7), src, dst, eattr, w.tb);
  k_elu_inplace<<<dim3((unsigned)cdiv((long)NN * 64, 256)), 256, 0, stream>>>(w.tb, (long)NN * 64);
  bn_stats(stream, w.tb, 64, w);
  bn_apply(stream, w.tb, w.xa, P(8), P(9), 64, 0, w);

  // ---- residual blocks
  const int chans[4] = {64, 128, 256, 512};
  for (int i = 0; i < 3; ++i) {
    int cin = chans[i], c = chans[i + 1];
    int pb = 10 + 15 * i;
    // h1 = elu(bn(conv_l1(x)))
    spline_conv(stream, w, w.xa, cin, c, P(pb + 0), P(pb + 1), P(pb + 2), src, dst, eattr, w.tb);
    bn_stats(stream, w.tb, c, w);
    bn_apply(stream, w.tb, w.hb, P(pb + 3), P(pb + 4), c, 1, w);
    // h = bn(conv_l2(h1))
    spline_conv(stream, w, w.hb, c, c, P(pb + 5), P(pb + 6), P(pb + 7), src, dst, eattr, w.tb);
    bn_stats(stream, w.tb, c, w);
    bn_apply(stream, w.tb, w.tb, P(pb + 8), P(pb + 9), c, 0, w);
    // s = bn(conv_sc(x))
    conv1x1(stream, w, w.xa, cin, c, P(pb + 10), P(pb + 11), P(pb + 12), src, dst, w.sb);
    bn_stats(stream, w.sb, c, w);
    bn_apply(stream, w.sb, w.sb, P(pb + 13), P(pb + 14), c, 0, w);
    // x = elu(h + s)
    k_add_elu<<<dim3((unsigned)cdiv((long)NN * c, 256)), 256, 0, stream>>>(w.tb, w.sb, w.xa, (long)NN * c);
  }

  // ---- voxel grid + max_pool_x
  k_seg<<<dim3((unsigned)cdiv(NN, 256)), 256, 0, stream>>>(pos, batch, w.seg);
  k_pool_max<<<dim3(NSEG), 256, 0, stream>>>(w.xa, w.seg, w.pooled, 512);

  // ---- fc1: elu(pooled_flat[8,32768] @ W[32768,1024] + b)  via WMMA
  k_cast_f32_f16<<<dim3((unsigned)cdiv((long)NGRAPH * FC1IN, 256)), 256, 0, stream>>>(
      w.pooled, w.pooledh, (long)NGRAPH * FC1IN);
  k_castT_pad<<<dim3((unsigned)cdiv((long)FC1IN * FC1OUT, 256)), 256, 0, stream>>>(
      P(55), w.Wh, FC1IN, FC1OUT, FC1IN, FC1OUT);         // Wt[1024][32768]
  gemm(stream, w.pooledh, w.Wh, w.fc1out, NGRAPH, FC1OUT, FC1IN, FC1IN, FC1IN, FC1OUT);
  k_bias_elu<<<dim3((unsigned)cdiv((long)NGRAPH * FC1OUT, 256)), 256, 0, stream>>>(
      w.fc1out, P(56), FC1OUT, (long)NGRAPH * FC1OUT);

  // ---- fc2 + log_softmax -> d_out [8,10]
  k_fc2_logsoftmax<<<dim3(NGRAPH), 32, 0, stream>>>(w.fc1out, P(57), P(58), (float*)d_out);
}
